// SpatialAttention_70274254897373
// MI455X (gfx1250) — compile-verified, matrix operand order checked by reading, not run on hardware
//
#include <hip/hip_runtime.h>
#include <hip/hip_bf16.h>

#define BB 4
#define TT 12
#define BT 48      // B*T
#define NN 512
#define DIMD 128
#define HH 4
#define DK 32

typedef __attribute__((ext_vector_type(16))) _Float16 v16h;
typedef __attribute__((ext_vector_type(8)))  _Float16 v8h;
typedef __attribute__((ext_vector_type(4)))  _Float16 v4h;
typedef __attribute__((ext_vector_type(8)))  float    v8f;
typedef __attribute__((ext_vector_type(4)))  float    v4f;
typedef __attribute__((ext_vector_type(4)))  int      v4i;

// ---- WMMA operand tile loaders (CDNA5 16x16x32 f16 layout) ----
// A-operand (16xK tile, K=32): lane L holds row (L&15); K elements
// {koff..koff+7, koff+16..koff+23} with koff = (L>=16)?8:0.
// The B-operand (32x16) has the lane-transposed layout, so loading
// B[k][n] from an [n][k] row-major array uses the *same* code with
// "row" = column index n.
__device__ inline v16h load_tile_f16(const _Float16* __restrict__ base,
                                     int row0, int k0, int ld, int lane) {
  int r = lane & 15;
  int koff = (lane >> 4) << 3;          // 0 or 8
  const _Float16* p = base + (size_t)(row0 + r) * ld + k0 + koff;
  v8h lo = *(const v8h*)p;
  v8h hi = *(const v8h*)(p + 16);
  v16h out{};
  for (int j = 0; j < 8; ++j) { out[j] = lo[j]; out[8 + j] = hi[j]; }
  return out;
}

__device__ inline v16h load_tile_f32(const float* __restrict__ base,
                                     int row0, int k0, int ld, int lane) {
  int r = lane & 15;
  int koff = (lane >> 4) << 3;
  const float* p = base + (size_t)(row0 + r) * ld + k0 + koff;
  v16h out{};
  for (int j = 0; j < 8; ++j) {
    out[j]     = (_Float16)p[j];
    out[8 + j] = (_Float16)p[j + 16];
  }
  return out;
}

// ---------------- Kernel 1: fused QKV projection ----------------
// y = x @ W^T + b ; split heads. Q,K -> f16 [bh][n][dk], V -> f16 [bh][dk][n]
__global__ __launch_bounds__(32)
void proj_qkv_kernel(const float* __restrict__ q_in, const float* __restrict__ k_in,
                     const float* __restrict__ v_in,
                     const float* __restrict__ Wq, const float* __restrict__ Wk,
                     const float* __restrict__ Wv,
                     const float* __restrict__ bq, const float* __restrict__ bk,
                     const float* __restrict__ bv,
                     _Float16* __restrict__ Qh, _Float16* __restrict__ Kh,
                     _Float16* __restrict__ Vt) {
  int lane = threadIdx.x;
  int tile = blockIdx.x;            // BT * (N/16)
  int e0   = blockIdx.y << 4;       // output feature tile
  int z    = blockIdx.z;            // 0=q 1=k 2=v
  int bt   = tile >> 5;             // N/16 == 32
  int n0   = (tile & 31) << 4;

  const float* x; const float* W; const float* bias;
  if (z == 0)      { x = q_in; W = Wq; bias = bq; }
  else if (z == 1) { x = k_in; W = Wk; bias = bk; }
  else             { x = v_in; W = Wv; bias = bv; }

  const float* xbt = x + (size_t)bt * NN * DIMD;
  v8f acc = {};
  for (int kc = 0; kc < DIMD; kc += 32) {
    v16h a = load_tile_f32(xbt, n0, kc, DIMD, lane);   // x[n][d]
    v16h b = load_tile_f32(W,  e0, kc, DIMD, lane);    // B[d][e] = W[e][d]
    acc = __builtin_amdgcn_wmma_f32_16x16x32_f16(false, a, false, b,
                                                 (short)0, acc, false, false);
  }
  int col   = lane & 15;
  int rbase = (lane >> 4) << 3;
  int e  = e0 + col;
  int h  = e >> 5;
  int dl = e & 31;
  float bia = bias[e];

  if (z < 2) {
    _Float16* dst = (z == 0 ? Qh : Kh) + (size_t)(bt * HH + h) * NN * DK;
    for (int i = 0; i < 8; ++i) {
      int n = n0 + rbase + i;
      dst[(size_t)n * DK + dl] = (_Float16)(acc[i] + bia);
    }
  } else {
    _Float16* dst = Vt + ((size_t)(bt * HH + h) * DK + dl) * NN + (n0 + rbase);
    v8h pk;
    for (int i = 0; i < 8; ++i) pk[i] = (_Float16)(acc[i] + bia);
    *(v8h*)dst = pk;                 // 8 consecutive n values -> b128 store
  }
}

// ---------------- Kernel 2: attention per (bt, h, 16-row tile) ----------------
__global__ __launch_bounds__(32)
void attn_kernel(const _Float16* __restrict__ Qh, const _Float16* __restrict__ Kh,
                 const _Float16* __restrict__ Vt,
                 const int* __restrict__ mask, const float* __restrict__ adjm,
                 _Float16* __restrict__ Xcat) {
  __shared__ float    sc[16][NN];     // raw scores (32 KB)
  __shared__ _Float16 at[16][NN];     // attention weights f16 (16 KB)

  int lane = threadIdx.x;
  int n0 = blockIdx.x << 4;
  int h  = blockIdx.y;
  int bt = blockIdx.z;
  int bh = bt * HH + h;
  const _Float16* Qb = Qh + (size_t)bh * NN * DK;
  const _Float16* Kb = Kh + (size_t)bh * NN * DK;
  const _Float16* Vb = Vt + (size_t)bh * DK * NN;

  int col   = lane & 15;
  int rbase = (lane >> 4) << 3;

  // Softmax-phase row/col split: 2 lanes per row, 256 cols each
  int r  = lane >> 1;
  int c0 = (lane & 1) << 8;
  const int*   mrow = mask + ((size_t)bt * NN + (n0 + r)) * NN + c0;
  const float* arow = adjm + ((size_t)bt * NN + (n0 + r)) * NN + c0;

  // Prefetch the dominant HBM stream (mask+adjm, ~100 MB total across the
  // grid) so it overlaps the QK^T WMMA phase. Lowers to global_prefetch_b8.
  for (int o = 0; o < 1024; o += 128) {
    __builtin_prefetch((const char*)mrow + o, 0, 1);
    __builtin_prefetch((const char*)arow + o, 0, 1);
  }

  // scores = Q K^T : 32 tiles of 16x16, K-dim = 32 (one WMMA each)
  v16h aQ = load_tile_f16(Qb, n0, 0, DK, lane);
  for (int mt = 0; mt < 32; ++mt) {
    v16h bK = load_tile_f16(Kb, mt << 4, 0, DK, lane);  // B[d][m] = K[m][d]
    v8f d = {};
    d = __builtin_amdgcn_wmma_f32_16x16x32_f16(false, aQ, false, bK,
                                               (short)0, d, false, false);
    int m = (mt << 4) + col;
    for (int i = 0; i < 8; ++i) sc[rbase + i][m] = d[i];
  }
  __syncthreads();

  // fused scale -> mask -> softmax -> *adjm, 4-wide vector passes
  const float scale = 0.1767766952966369f;   // 1/sqrt(32)
  const v4i* mrow4 = (const v4i*)mrow;
  const v4f* arow4 = (const v4f*)arow;
  v4f* scrow4 = (v4f*)&sc[r][c0];
  v4h* atrow4 = (v4h*)&at[r][c0];

  float mx = -3.0e38f;
  for (int c4 = 0; c4 < 64; ++c4) {
    v4i mk = mrow4[c4];
    v4f s  = scrow4[c4];
    for (int j = 0; j < 4; ++j) {
      float sv = s[j] * scale;
      if (mk[j] != 0) sv = -1.0e9f;
      s[j] = sv;
      mx = fmaxf(mx, sv);
    }
    scrow4[c4] = s;
  }
  mx = fmaxf(mx, __shfl_xor(mx, 1, 32));

  float sum = 0.f;
  for (int c4 = 0; c4 < 64; ++c4) {
    v4f s = scrow4[c4];
    for (int j = 0; j < 4; ++j) {
      float e = __expf(s[j] - mx);
      s[j] = e;
      sum += e;
    }
    scrow4[c4] = s;
  }
  sum += __shfl_xor(sum, 1, 32);
  float inv = 1.0f / sum;

  for (int c4 = 0; c4 < 64; ++c4) {
    v4f s = scrow4[c4];
    v4f a = arow4[c4];
    v4h o;
    for (int j = 0; j < 4; ++j) o[j] = (_Float16)(s[j] * inv * a[j]);
    atrow4[c4] = o;
  }
  __syncthreads();

  // x = attn @ V : K-loop over 512 in chunks of 32, two 16-wide d tiles
  v8f acc0 = {}, acc1 = {};
  for (int mt = 0; mt < 16; ++mt) {
    int k0 = mt << 5;
    int rr = lane & 15;
    const _Float16* ap = &at[rr][k0 + rbase];
    v8h lo = *(const v8h*)ap;
    v8h hi = *(const v8h*)(ap + 16);
    v16h aA{};
    for (int j = 0; j < 8; ++j) { aA[j] = lo[j]; aA[8 + j] = hi[j]; }
    v16h b0 = load_tile_f16(Vb, 0,  k0, NN, lane);   // B[m][d] = Vt[d][m]
    v16h b1 = load_tile_f16(Vb, 16, k0, NN, lane);
    acc0 = __builtin_amdgcn_wmma_f32_16x16x32_f16(false, aA, false, b0,
                                                  (short)0, acc0, false, false);
    acc1 = __builtin_amdgcn_wmma_f32_16x16x32_f16(false, aA, false, b1,
                                                  (short)0, acc1, false, false);
  }
  // concat heads: Xcat[bt][n][h*32 + d]
  _Float16* xb = Xcat + (size_t)bt * NN * DIMD + h * 32;
  for (int i = 0; i < 8; ++i) {
    int n = n0 + rbase + i;
    xb[(size_t)n * DIMD + col]      = (_Float16)acc0[i];
    xb[(size_t)n * DIMD + 16 + col] = (_Float16)acc1[i];
  }
}

// ---------------- Kernel 3: output projection ----------------
__global__ __launch_bounds__(32)
void out_proj_kernel(const _Float16* __restrict__ Xcat,
                     const float* __restrict__ Wp, const float* __restrict__ bp,
                     float* __restrict__ out) {
  int lane = threadIdx.x;
  int tile = blockIdx.x;
  int e0   = blockIdx.y << 4;
  int bt   = tile >> 5;
  int n0   = (tile & 31) << 4;
  const _Float16* xb = Xcat + (size_t)bt * NN * DIMD;

  v8f acc = {};
  for (int kc = 0; kc < DIMD; kc += 32) {
    v16h a = load_tile_f16(xb, n0, kc, DIMD, lane);
    v16h b = load_tile_f32(Wp, e0, kc, DIMD, lane);   // B[d][e] = Wp[e][d]
    acc = __builtin_amdgcn_wmma_f32_16x16x32_f16(false, a, false, b,
                                                 (short)0, acc, false, false);
  }
  int col   = lane & 15;
  int rbase = (lane >> 4) << 3;
  float bia = bp[e0 + col];
  float* ob = out + (size_t)bt * NN * DIMD;
  for (int i = 0; i < 8; ++i)
    ob[(size_t)(n0 + rbase + i) * DIMD + e0 + col] = acc[i] + bia;
}

extern "C" void kernel_launch(void* const* d_in, const int* in_sizes, int n_in,
                              void* d_out, int out_size, void* d_ws, size_t ws_size,
                              hipStream_t stream) {
  const float* query = (const float*)d_in[0];
  const float* key_  = (const float*)d_in[1];
  const float* value = (const float*)d_in[2];
  const int*   mask  = (const int*)  d_in[3];
  const float* adjm  = (const float*)d_in[4];
  const float* Wq = (const float*)d_in[5];  const float* bq = (const float*)d_in[6];
  const float* Wk = (const float*)d_in[7];  const float* bk = (const float*)d_in[8];
  const float* Wv = (const float*)d_in[9];  const float* bv = (const float*)d_in[10];
  const float* Wp = (const float*)d_in[11]; const float* bp = (const float*)d_in[12];

  size_t qkv_elems = (size_t)BT * HH * NN * DK;   // 3,145,728 f16 each
  _Float16* Qh   = (_Float16*)d_ws;
  _Float16* Kh   = Qh + qkv_elems;
  _Float16* Vt   = Kh + qkv_elems;
  _Float16* Xcat = Vt + qkv_elems;                // [bt][n][dim] f16

  dim3 g1(BT * (NN / 16), DIMD / 16, 3);
  proj_qkv_kernel<<<g1, 32, 0, stream>>>(query, key_, value, Wq, Wk, Wv,
                                         bq, bk, bv, Qh, Kh, Vt);

  dim3 g2(NN / 16, HH, BT);
  attn_kernel<<<g2, 32, 0, stream>>>(Qh, Kh, Vt, mask, adjm, Xcat);

  dim3 g3(BT * (NN / 16), DIMD / 16);
  out_proj_kernel<<<g3, 32, 0, stream>>>(Xcat, Wp, bp, (float*)d_out);
}
